// gvae_65249143160988
// MI455X (gfx1250) — matrix operand order repaired and checked by reading, not compiled, not run on hardware
//
#include <hip/hip_runtime.h>
#include <hip/hip_bf16.h>

// ---------------------------------------------------------------------------
// Fused Graph-VAE for MI455X (gfx1250, wave32, WMMA f16 16x16x32).
// Edge path, node path and decoders fully fused: e_lat / z_e / edge_in never
// touch HBM. Segment sums via L2 f32 atomics (per-node) and LDS accumulators
// (per-graph). Weights pre-transcoded to zero-padded 32x16 f16 tiles so K/N
// padding contributes exact zeros to the WMMAs. All per-lane bias scalars are
// hoisted out of the tile loop; SELU and the gathers are branchless (cndmask).
// ---------------------------------------------------------------------------

typedef __attribute__((ext_vector_type(16))) _Float16 v16h;
typedef __attribute__((ext_vector_type(8)))  _Float16 v8h;
typedef __attribute__((ext_vector_type(8)))  float    v8f;

#define ROWSTRIDE  40   // LDS stage row pitch in halves (80B -> conflict-free b128)
#define TILE_HALFS 512  // one 32x16 f16 weight tile

// Branchless SELU: exp() is evaluated unconditionally on a clamped argument so
// the select lowers to v_cndmask instead of exec-mask branches.
__device__ __forceinline__ float selu_f(float x) {
  const float kS = 1.0507009873554805f, kA = 1.6732632423543772f;
  const float e = __expf(fminf(x, 0.f));
  const float neg = kS * kA * (e - 1.f);
  return x > 0.f ? kS * x : neg;
}

__device__ __forceinline__ v8f wmma16(v16h a, v16h b, v8f c) {
  return __builtin_amdgcn_wmma_f32_16x16x32_f16(false, a, false, b, (short)0, c,
                                                false, false);
}

// A fragment (16x32 f16): lane holds row m=lane&15; lanes 0-15 carry K 0..15,
// lanes 16-31 carry K 16..31 (contiguous, matching load_B).
__device__ __forceinline__ v16h load_A(const _Float16* st, int lane) {
  const int m = lane & 15, sel = lane >> 4;
  const _Float16* row = st + m * ROWSTRIDE + sel * 16;
  v8h lo = *(const v8h*)(row);        // 16B aligned (pitch 80B, off 0/32B)
  v8h hi = *(const v8h*)(row + 8);
  return __builtin_shufflevector(lo, hi, 0, 1, 2, 3, 4, 5, 6, 7,
                                 8, 9, 10, 11, 12, 13, 14, 15);
}

// B fragment (32x16 f16) from a row-major padded [32][16] tile in global ws:
// lane holds column n=lane&15; lanes 0-15 rows 0..15, lanes 16-31 rows 16..31.
__device__ __forceinline__ v16h load_B(const _Float16* w, int lane) {
  const int n = lane & 15, sel = lane >> 4;
  v16h b;
#pragma unroll
  for (int i = 0; i < 16; ++i) b[i] = w[(sel * 16 + i) * 16 + n];
  return b;
}

// D tile (C/D layout: lane holds col n=lane&15, rows m=r+8*(lane>=16)) ->
// +bias (+selu) -> f16 into LDS stage; cols >= nvalid written as exact zero.
// bn is this lane's pre-hoisted bias scalar for column n (0 when n>=nvalid).
__device__ __forceinline__ void stage_D(_Float16* st, v8f d, float bn,
                                        int nvalid, int ntile, bool act, int lane) {
  const int sel = lane >> 4, n = (lane & 15) + 16 * ntile;
  const bool ok = n < nvalid;
#pragma unroll
  for (int r = 0; r < 8; ++r) {
    float v = d[r] + bn;
    if (act) v = selu_f(v);
    v = ok ? v : 0.f;
    st[(r + 8 * sel) * ROWSTRIDE + n] = (_Float16)v;
  }
}

__device__ __forceinline__ void stage_zero(_Float16* st, int ntile, int lane) {
  const int sel = lane >> 4, n = (lane & 15) + 16 * ntile;
#pragma unroll
  for (int r = 0; r < 8; ++r)
    st[(r + 8 * sel) * ROWSTRIDE + n] = (_Float16)0.f;
}

__device__ __forceinline__ float bias_at(const float* b, int idx, int nvalid) {
  return idx < nvalid ? b[idx] : 0.f;
}

// ---------------------------------------------------------------------------
struct EdgeParams {
  const float *edges, *nodes, *eps_e;
  const int *senders, *receivers, *node_graph_ids;
  const _Float16* Wt;
  const float *b_e1, *b_e2, *b_e3, *b_fc3, *b_fc4, *b_ed1, *b_ed2;
  float *agg_e_node, *agg_e_g;
  float *out_e_dec, *out_mu_e, *out_lv_e;
  int E;
};

__global__ __launch_bounds__(256) void edge_kernel(EdgeParams p) {
  __shared__ _Float16 stage[8][16 * ROWSTRIDE];
  __shared__ int recvIdx[8][16];
  __shared__ int gidIdx[8][16];
  __shared__ float gacc[64 * 10];

  const int tid = threadIdx.x, wave = tid >> 5, lane = tid & 31;
  for (int i = tid; i < 640; i += 256) gacc[i] = 0.f;
  __syncthreads();

  const _Float16* Wt = p.Wt;
  const v16h B_e1a = load_B(Wt + 0 * TILE_HALFS, lane);
  const v16h B_e1b = load_B(Wt + 1 * TILE_HALFS, lane);
  const v16h B_e2a = load_B(Wt + 2 * TILE_HALFS, lane);
  const v16h B_e2b = load_B(Wt + 3 * TILE_HALFS, lane);
  const v16h B_e3  = load_B(Wt + 4 * TILE_HALFS, lane);
  const v16h B_m3a = load_B(Wt + 5 * TILE_HALFS, lane);
  const v16h B_m3b = load_B(Wt + 6 * TILE_HALFS, lane);
  const v16h B_v4a = load_B(Wt + 7 * TILE_HALFS, lane);
  const v16h B_v4b = load_B(Wt + 8 * TILE_HALFS, lane);
  const v16h B_d1  = load_B(Wt + 9 * TILE_HALFS, lane);
  const v16h B_d2  = load_B(Wt + 10 * TILE_HALFS, lane);

  // per-lane bias scalars (each lane only ever touches columns n0 and n0+16)
  const int n0 = lane & 15;
  const float be1_0 = bias_at(p.b_e1, n0, 25), be1_1 = bias_at(p.b_e1, n0 + 16, 25);
  const float be2_0 = bias_at(p.b_e2, n0, 20), be2_1 = bias_at(p.b_e2, n0 + 16, 20);
  const float be3_0 = bias_at(p.b_e3, n0, 10);
  const float bm_0  = bias_at(p.b_fc3, n0, 20), bm_1 = bias_at(p.b_fc3, n0 + 16, 20);
  const float bv_0  = bias_at(p.b_fc4, n0, 20), bv_1 = bias_at(p.b_fc4, n0 + 16, 20);
  const float bd1_0 = bias_at(p.b_ed1, n0, 10);
  const float bd2_0 = bias_at(p.b_ed2, n0, 5);

  _Float16* st = stage[wave];
  const int m = lane & 15, sel = lane >> 4;
  const int numTiles = p.E >> 4;              // E divisible by 16 (3.2M)
  const int waveCount = gridDim.x * 8;
  const v8f zeroC = {};

  for (int t = blockIdx.x * 8 + wave; t < numTiles; t += waveCount) {
    const int e0 = t << 4;
    const int e = e0 + m;
    // ---- build edge_in tile [16 x 31] (+zero pad col 31), branchless ----
    const int recv = p.receivers[e];
    const int send = p.senders[e];
    if (sel == 0) { recvIdx[wave][m] = recv; gidIdx[wave][m] = p.node_graph_ids[recv]; }
    const float* er = p.edges + (long)e * 5;
    const float* nr = p.nodes + (long)recv * 13;   // nodes table: L2-resident
    const float* ns = p.nodes + (long)send * 13;
#pragma unroll
    for (int j = 0; j < 16; ++j) {
      const int c = sel * 16 + j;
      const int cc = c < 31 ? c : 30;  // clamp: pad column reads a valid addr
      const float* ptr = cc < 5 ? (er + cc)
                                : (cc < 18 ? (nr + cc - 5) : (ns + cc - 18));
      float v = *ptr;
      v = c < 31 ? v : 0.f;
      st[m * ROWSTRIDE + c] = (_Float16)v;
    }
    // ---- edge MLP: 31 -> 25 (selu) -> 20 (selu) -> 10 ----
    v16h a = load_A(st, lane);
    v8f d0 = wmma16(a, B_e1a, zeroC);
    v8f d1 = wmma16(a, B_e1b, zeroC);
    stage_D(st, d0, be1_0, 25, 0, true, lane);
    stage_D(st, d1, be1_1, 25, 1, true, lane);
    a = load_A(st, lane);
    d0 = wmma16(a, B_e2a, zeroC);
    d1 = wmma16(a, B_e2b, zeroC);
    stage_D(st, d0, be2_0, 20, 0, true, lane);
    stage_D(st, d1, be2_1, 20, 1, true, lane);
    a = load_A(st, lane);
    const v8f dl = wmma16(a, B_e3, zeroC);          // e_lat (cols 0..9)
    // ---- e_lat: stage for fc3/fc4 + segment sums ----
    stage_D(st, dl, be3_0, 10, 0, false, lane);
    stage_zero(st, 1, lane);
    if (n0 < 10) {
#pragma unroll
      for (int r = 0; r < 8; ++r) {
        const int mm = r + 8 * sel;
        const float v = dl[r] + be3_0;
        unsafeAtomicAdd(&p.agg_e_node[(long)recvIdx[wave][mm] * 10 + n0], v);
        atomicAdd(&gacc[gidIdx[wave][mm] * 10 + n0], v);  // ds_add_f32
      }
    }
    // ---- VAE heads: mu_e / logvar_e / z_e (z staged f16 for decoder) ----
    a = load_A(st, lane);
    const v8f mu0 = wmma16(a, B_m3a, zeroC);
    const v8f mu1 = wmma16(a, B_m3b, zeroC);
    const v8f lv0 = wmma16(a, B_v4a, zeroC);
    const v8f lv1 = wmma16(a, B_v4b, zeroC);
#pragma unroll
    for (int nt = 0; nt < 2; ++nt) {
      const v8f& dm = nt ? mu1 : mu0;
      const v8f& dv = nt ? lv1 : lv0;
      const float bm = nt ? bm_1 : bm_0;
      const float bv = nt ? bv_1 : bv_0;
      const int n = n0 + 16 * nt;
      const int ncl = n < 20 ? n : 19;              // safe eps column
      float muv[8], lvv[8];
      _Float16 zv[8];
#pragma unroll
      for (int r = 0; r < 8; ++r) {
        const long ei = (long)(e0 + r + 8 * sel);
        muv[r] = dm[r] + bm;
        lvv[r] = dv[r] + bv;
        const float z = muv[r] + p.eps_e[ei * 20 + ncl] * __expf(0.5f * lvv[r]);
        zv[r] = (_Float16)(n < 20 ? z : 0.f);
      }
      if (n < 20) {                                 // single predicated block
#pragma unroll
        for (int r = 0; r < 8; ++r) {
          const long ei = (long)(e0 + r + 8 * sel);
          p.out_mu_e[ei * 20 + n] = muv[r];         // coalesced per row
          p.out_lv_e[ei * 20 + n] = lvv[r];
        }
      }
#pragma unroll
      for (int r = 0; r < 8; ++r)
        st[(r + 8 * sel) * ROWSTRIDE + n] = zv[r];
    }
    // ---- edge decoder: 20 -> 10 (selu) -> 5 ----
    a = load_A(st, lane);
    d0 = wmma16(a, B_d1, zeroC);
    stage_D(st, d0, bd1_0, 10, 0, true, lane);
    stage_zero(st, 1, lane);
    a = load_A(st, lane);
    d0 = wmma16(a, B_d2, zeroC);
    if (n0 < 5) {
#pragma unroll
      for (int r = 0; r < 8; ++r)
        p.out_e_dec[(long)(e0 + r + 8 * sel) * 5 + n0] = d0[r] + bd2_0;
    }
  }
  __syncthreads();
  for (int i = tid; i < 640; i += 256)
    unsafeAtomicAdd(&p.agg_e_g[i], gacc[i]);
}

// ---------------------------------------------------------------------------
struct NodeParams {
  const float *nodes, *agg_e_node, *eps_n;
  const int* node_graph_ids;
  const _Float16* Wt;
  const float *b_n1, *b_n2, *b_fc1, *b_fc2, *b_nd1, *b_nd2;
  float* agg_n_g;
  float *out_n_dec, *out_mu_n, *out_lv_n;
  int N;
};

__global__ __launch_bounds__(256) void node_kernel(NodeParams p) {
  __shared__ _Float16 stage[8][16 * ROWSTRIDE];
  __shared__ int gidIdx[8][16];
  __shared__ float gacc[64 * 10];

  const int tid = threadIdx.x, wave = tid >> 5, lane = tid & 31;
  for (int i = tid; i < 640; i += 256) gacc[i] = 0.f;
  __syncthreads();

  const _Float16* Wt = p.Wt;
  const v16h B_n1a = load_B(Wt + 11 * TILE_HALFS, lane);
  const v16h B_n1b = load_B(Wt + 12 * TILE_HALFS, lane);
  const v16h B_n2  = load_B(Wt + 13 * TILE_HALFS, lane);
  const v16h B_f1a = load_B(Wt + 14 * TILE_HALFS, lane);
  const v16h B_f1b = load_B(Wt + 15 * TILE_HALFS, lane);
  const v16h B_f2a = load_B(Wt + 16 * TILE_HALFS, lane);
  const v16h B_f2b = load_B(Wt + 17 * TILE_HALFS, lane);
  const v16h B_d1  = load_B(Wt + 18 * TILE_HALFS, lane);
  const v16h B_d2  = load_B(Wt + 19 * TILE_HALFS, lane);

  const int n0 = lane & 15;
  const float bn1_0 = bias_at(p.b_n1, n0, 18), bn1_1 = bias_at(p.b_n1, n0 + 16, 18);
  const float bn2_0 = bias_at(p.b_n2, n0, 10);
  const float bf1_0 = bias_at(p.b_fc1, n0, 20), bf1_1 = bias_at(p.b_fc1, n0 + 16, 20);
  const float bf2_0 = bias_at(p.b_fc2, n0, 20), bf2_1 = bias_at(p.b_fc2, n0 + 16, 20);
  const float bnd1_0 = bias_at(p.b_nd1, n0, 16);
  const float bnd2_0 = bias_at(p.b_nd2, n0, 13);

  _Float16* st = stage[wave];
  const int m = lane & 15, sel = lane >> 4;
  const int numTiles = p.N >> 4;              // N divisible by 16 (100K)
  const int waveCount = gridDim.x * 8;
  const v8f zeroC = {};

  for (int t = blockIdx.x * 8 + wave; t < numTiles; t += waveCount) {
    const int v0 = (t << 4) + m;
    if (sel == 0) gidIdx[wave][m] = p.node_graph_ids[v0];
    // node_in = [agg_e_node(10) | nodes(13) | 0-pad], branchless gather
    const float* ag = p.agg_e_node + (long)v0 * 10;
    const float* nd = p.nodes + (long)v0 * 13;
#pragma unroll
    for (int j = 0; j < 16; ++j) {
      const int c = sel * 16 + j;
      const int cc = c < 23 ? c : 22;
      const float* ptr = cc < 10 ? (ag + cc) : (nd + cc - 10);
      float v = *ptr;
      v = c < 23 ? v : 0.f;
      st[m * ROWSTRIDE + c] = (_Float16)v;
    }
    // ---- node MLP: 23 -> 18 (selu) -> 10 ----
    v16h a = load_A(st, lane);
    v8f d0 = wmma16(a, B_n1a, zeroC);
    v8f d1 = wmma16(a, B_n1b, zeroC);
    stage_D(st, d0, bn1_0, 18, 0, true, lane);
    stage_D(st, d1, bn1_1, 18, 1, true, lane);
    a = load_A(st, lane);
    const v8f dl = wmma16(a, B_n2, zeroC);          // n_lat (cols 0..9)
    stage_D(st, dl, bn2_0, 10, 0, false, lane);
    stage_zero(st, 1, lane);
    if (n0 < 10) {
#pragma unroll
      for (int r = 0; r < 8; ++r)
        atomicAdd(&gacc[gidIdx[wave][r + 8 * sel] * 10 + n0], dl[r] + bn2_0);
    }
    // ---- VAE heads: mu_n / logvar_n / z_n ----
    a = load_A(st, lane);
    const v8f mu0 = wmma16(a, B_f1a, zeroC);
    const v8f mu1 = wmma16(a, B_f1b, zeroC);
    const v8f lv0 = wmma16(a, B_f2a, zeroC);
    const v8f lv1 = wmma16(a, B_f2b, zeroC);
#pragma unroll
    for (int nt = 0; nt < 2; ++nt) {
      const v8f& dm = nt ? mu1 : mu0;
      const v8f& dv = nt ? lv1 : lv0;
      const float bm = nt ? bf1_1 : bf1_0;
      const float bv = nt ? bf2_1 : bf2_0;
      const int n = n0 + 16 * nt;
      const int ncl = n < 20 ? n : 19;
      float muv[8], lvv[8];
      _Float16 zv[8];
#pragma unroll
      for (int r = 0; r < 8; ++r) {
        const long vi = (long)((t << 4) + r + 8 * sel);
        muv[r] = dm[r] + bm;
        lvv[r] = dv[r] + bv;
        const float z = muv[r] + p.eps_n[vi * 20 + ncl] * __expf(0.5f * lvv[r]);
        zv[r] = (_Float16)(n < 20 ? z : 0.f);
      }
      if (n < 20) {
#pragma unroll
        for (int r = 0; r < 8; ++r) {
          const long vi = (long)((t << 4) + r + 8 * sel);
          p.out_mu_n[vi * 20 + n] = muv[r];
          p.out_lv_n[vi * 20 + n] = lvv[r];
        }
      }
#pragma unroll
      for (int r = 0; r < 8; ++r)
        st[(r + 8 * sel) * ROWSTRIDE + n] = zv[r];
    }
    // ---- node decoder: 20 -> 16 (selu) -> 13 ----
    a = load_A(st, lane);
    d0 = wmma16(a, B_d1, zeroC);
    stage_D(st, d0, bnd1_0, 16, 0, true, lane);
    stage_zero(st, 1, lane);
    a = load_A(st, lane);
    d0 = wmma16(a, B_d2, zeroC);
    if (n0 < 13) {
#pragma unroll
      for (int r = 0; r < 8; ++r)
        p.out_n_dec[(long)((t << 4) + r + 8 * sel) * 13 + n0] = d0[r] + bnd2_0;
    }
  }
  __syncthreads();
  for (int i = tid; i < 640; i += 256)
    unsafeAtomicAdd(&p.agg_n_g[i], gacc[i]);
}

// ---------------------------------------------------------------------------
__global__ void global_kernel(const float* agg_e_g, const float* agg_n_g,
                              const float* W0, const float* b0,
                              const float* W1, const float* b1,
                              const float* W2, const float* b2, float* g_out) {
  const int g = threadIdx.x;
  if (g >= 64) return;
  float in[20], h0[15], h1[15];
  for (int i = 0; i < 10; ++i) { in[i] = agg_e_g[g * 10 + i]; in[10 + i] = agg_n_g[g * 10 + i]; }
  for (int o = 0; o < 15; ++o) {
    float s = b0[o];
    for (int i = 0; i < 20; ++i) s += in[i] * W0[i * 15 + o];
    h0[o] = selu_f(s);
  }
  for (int o = 0; o < 15; ++o) {
    float s = b1[o];
    for (int i = 0; i < 15; ++i) s += h0[i] * W1[i * 15 + o];
    h1[o] = selu_f(s);
  }
  for (int o = 0; o < 10; ++o) {
    float s = b2[o];
    for (int i = 0; i < 15; ++i) s += h1[i] * W2[i * 15 + o];
    g_out[g * 10 + o] = s;
  }
}

__global__ void zero_kernel(float* p, long n) {
  long i = (long)blockIdx.x * blockDim.x + threadIdx.x;
  const long stride = (long)gridDim.x * blockDim.x;
  for (; i < n; i += stride) p[i] = 0.f;
}

// Transcode f32 [K][Nout] weights into ntiles zero-padded 32x16 f16 tiles.
__global__ void prep_w_kernel(_Float16* dst, const float* W, int K, int Nout,
                              int ntiles) {
  const int total = ntiles * 512;
  for (int idx = blockIdx.x * blockDim.x + threadIdx.x; idx < total;
       idx += gridDim.x * blockDim.x) {
    const int tIdx = idx >> 9, pos = idx & 511;
    const int k = pos >> 4, n = (pos & 15) + tIdx * 16;
    dst[idx] = (_Float16)((k < K && n < Nout) ? W[k * Nout + n] : 0.f);
  }
}

// ---------------------------------------------------------------------------
extern "C" void kernel_launch(void* const* d_in, const int* in_sizes, int n_in,
                              void* d_out, int out_size, void* d_ws, size_t ws_size,
                              hipStream_t stream) {
  const float* nodes          = (const float*)d_in[0];
  const float* edges          = (const float*)d_in[1];
  const int*   senders        = (const int*)d_in[2];
  const int*   receivers      = (const int*)d_in[3];
  const int*   node_graph_ids = (const int*)d_in[4];
  const float* eps_n          = (const float*)d_in[5];
  const float* eps_e          = (const float*)d_in[6];
  const int N = in_sizes[0] / 13;
  const int E = in_sizes[1] / 5;

  // workspace: [agg_e_node N*10 f32][agg_e_g 640][agg_n_g 640][20 f16 tiles]
  float* agg_e_node = (float*)d_ws;
  float* agg_e_g    = agg_e_node + (size_t)N * 10;
  float* agg_n_g    = agg_e_g + 640;
  _Float16* Wt      = (_Float16*)(agg_n_g + 640);

  // output tuple, flat: n_dec, e_dec, g_out, mu_n, logvar_n, mu_e, logvar_e
  float* out       = (float*)d_out;
  float* out_n_dec = out;
  float* out_e_dec = out_n_dec + (size_t)N * 13;
  float* out_g     = out_e_dec + (size_t)E * 5;
  float* out_mu_n  = out_g + 640;
  float* out_lv_n  = out_mu_n + (size_t)N * 20;
  float* out_mu_e  = out_lv_n + (size_t)N * 20;
  float* out_lv_e  = out_mu_e + (size_t)E * 20;

  zero_kernel<<<512, 256, 0, stream>>>(agg_e_node, (long)N * 10 + 1280);

  // params leaves (setup_inputs dict/list order)
  struct Plan { int din, base, K, Nout, ntiles; };
  const Plan plan[13] = {
      {7,  0, 31, 25, 2},  // edge_mlp W0
      {9,  2, 25, 20, 2},  // edge_mlp W1
      {11, 4, 20, 10, 1},  // edge_mlp W2
      {27, 5, 10, 20, 2},  // fc3 (mu_e)
      {29, 7, 10, 20, 2},  // fc4 (logvar_e)
      {31, 9, 20, 10, 1},  // edge_dec W0
      {33, 10, 10, 5, 1},  // edge_dec W1
      {13, 11, 23, 18, 2}, // node_mlp W0
      {15, 13, 18, 10, 1}, // node_mlp W1
      {23, 14, 10, 20, 2}, // fc1 (mu_n)
      {25, 16, 10, 20, 2}, // fc2 (logvar_n)
      {35, 18, 20, 16, 1}, // node_dec W0
      {37, 19, 16, 13, 1}, // node_dec W1
  };
  for (int i = 0; i < 13; ++i)
    prep_w_kernel<<<2, 256, 0, stream>>>(Wt + (size_t)plan[i].base * TILE_HALFS,
                                         (const float*)d_in[plan[i].din],
                                         plan[i].K, plan[i].Nout, plan[i].ntiles);

  EdgeParams ep;
  ep.edges = edges; ep.nodes = nodes; ep.eps_e = eps_e;
  ep.senders = senders; ep.receivers = receivers; ep.node_graph_ids = node_graph_ids;
  ep.Wt = Wt;
  ep.b_e1 = (const float*)d_in[8];  ep.b_e2 = (const float*)d_in[10];
  ep.b_e3 = (const float*)d_in[12]; ep.b_fc3 = (const float*)d_in[28];
  ep.b_fc4 = (const float*)d_in[30]; ep.b_ed1 = (const float*)d_in[32];
  ep.b_ed2 = (const float*)d_in[34];
  ep.agg_e_node = agg_e_node; ep.agg_e_g = agg_e_g;
  ep.out_e_dec = out_e_dec; ep.out_mu_e = out_mu_e; ep.out_lv_e = out_lv_e;
  ep.E = E;
  edge_kernel<<<1024, 256, 0, stream>>>(ep);

  NodeParams np;
  np.nodes = nodes; np.agg_e_node = agg_e_node; np.eps_n = eps_n;
  np.node_graph_ids = node_graph_ids; np.Wt = Wt;
  np.b_n1 = (const float*)d_in[14]; np.b_n2 = (const float*)d_in[16];
  np.b_fc1 = (const float*)d_in[24]; np.b_fc2 = (const float*)d_in[26];
  np.b_nd1 = (const float*)d_in[36]; np.b_nd2 = (const float*)d_in[38];
  np.agg_n_g = agg_n_g;
  np.out_n_dec = out_n_dec; np.out_mu_n = out_mu_n; np.out_lv_n = out_lv_n;
  np.N = N;
  node_kernel<<<256, 256, 0, stream>>>(np);

  global_kernel<<<1, 64, 0, stream>>>(agg_e_g, agg_n_g,
                                      (const float*)d_in[17], (const float*)d_in[18],
                                      (const float*)d_in[19], (const float*)d_in[20],
                                      (const float*)d_in[21], (const float*)d_in[22],
                                      out_g);
}
